// ScaledDotProductAttention_22789096472790
// MI455X (gfx1250) — matrix operand order repaired and checked by reading, not compiled, not run on hardware
//
#include <hip/hip_runtime.h>
#include <math.h>

typedef __attribute__((ext_vector_type(16))) _Float16 v16h;
typedef __attribute__((ext_vector_type(4)))  _Float16 v4h;
typedef __attribute__((ext_vector_type(8)))  float    v8f;
typedef unsigned int u32x4 __attribute__((ext_vector_type(4)));
typedef int          i32x8 __attribute__((ext_vector_type(8)));
typedef int          i32x4 __attribute__((ext_vector_type(4)));

// Problem constants (from reference): B=4,H=16,S=2048,DK=DV=128
#define B_    4
#define H_    16
#define S_    2048
#define DK_   128
#define DV_   128
#define QTILE 128   // Q rows per block (8 waves x 16 rows)
#define KCH   32    // KV rows per inner iteration (matches WMMA K=32)
#define NWAVES 8

__device__ __forceinline__ float red_max16(float v) {
    v = fmaxf(v, __shfl_xor(v, 1, 32));
    v = fmaxf(v, __shfl_xor(v, 2, 32));
    v = fmaxf(v, __shfl_xor(v, 4, 32));
    v = fmaxf(v, __shfl_xor(v, 8, 32));
    return v;
}
__device__ __forceinline__ float red_sum16(float v) {
    v += __shfl_xor(v, 1, 32);
    v += __shfl_xor(v, 2, 32);
    v += __shfl_xor(v, 4, 32);
    v += __shfl_xor(v, 8, 32);
    return v;
}

__device__ __forceinline__ v8f wmma_f16(v16h a, v16h b, v8f c) {
    // (neg_a, A, neg_b, B, c_mod, C, reuse_a, reuse_b)
    return __builtin_amdgcn_wmma_f32_16x16x32_f16(false, a, false, b,
                                                  (short)0, c, false, false);
}

// ---- Tensor Data Mover: generic 2-D tile global -> LDS ----
// Descriptor layout per CDNA5 ISA ch.8 (D# groups 0/1; groups 2/3 zero for 2D).
__device__ __forceinline__ void tdm_load_2d(unsigned lds_off, const void* gptr,
                                            unsigned dsize_code,   // 0:1B 1:2B 2:4B
                                            unsigned td0, unsigned td1,
                                            unsigned til0, unsigned til1,
                                            unsigned stride0) {
    unsigned long long ga = (unsigned long long)(size_t)gptr;
    u32x4 g0;
    g0[0] = 1u;                                     // count=1 (valid), user mode
    g0[1] = lds_off;                                // lds_addr (bytes)
    g0[2] = (unsigned)(ga & 0xffffffffu);           // global_addr[31:0]
    g0[3] = (unsigned)((ga >> 32) & 0x01ffffffu)    // global_addr[56:32]
          | 0x80000000u;                            // type=2 ("image")
    i32x8 g1;
    g1[0] = (int)(dsize_code << 16);                        // wg_mask=0 | data_size
    g1[1] = (int)((td0 & 0xffffu) << 16);                   // tensor_dim0 lo16
    g1[2] = (int)(((td0 >> 16) & 0xffffu) | ((td1 & 0xffffu) << 16));
    g1[3] = (int)(((td1 >> 16) & 0xffffu) | (til0 << 16));  // tile_dim0
    g1[4] = (int)(til1 & 0xffffu);                          // tile_dim1 | tile_dim2=0
    g1[5] = (int)stride0;                                   // tensor_dim0_stride lo32
    g1[6] = 0;
    g1[7] = 0;
    i32x4 z4 = {0, 0, 0, 0};
    i32x8 z8 = {0, 0, 0, 0, 0, 0, 0, 0};
    __builtin_amdgcn_tensor_load_to_lds(g0, g1, z4, z4, z8, 0);
}

// ======================= prepass kernels =======================
// K: f32 -> f16 elementwise (same layout)
__global__ __launch_bounds__(256)
void cvt_k_f16_kernel(const float* __restrict__ in, _Float16* __restrict__ out) {
    int i = blockIdx.x * 256 + threadIdx.x;     // one float4 per thread
    float4 f = ((const float4*)in)[i];
    v4h h;
    h[0] = (_Float16)f.x; h[1] = (_Float16)f.y;
    h[2] = (_Float16)f.z; h[3] = (_Float16)f.w;
    ((v4h*)out)[i] = h;
}

// V: f32 [bh][s][d] -> f16 transposed [bh][d][s] via 32x32 LDS tiles
__global__ __launch_bounds__(256)
void transpose_v_f16_kernel(const float* __restrict__ V, _Float16* __restrict__ Vt) {
    __shared__ float tile[32][33];
    const int bh = blockIdx.z;
    const int s0 = blockIdx.x * 32, d0 = blockIdx.y * 32;
    const int tx = threadIdx.x, ty = threadIdx.y;   // 32 x 8
    const float* src = V + ((size_t)bh * S_ + s0) * DV_ + d0;
    #pragma unroll
    for (int i = 0; i < 4; ++i)
        tile[ty + 8 * i][tx] = src[(size_t)(ty + 8 * i) * DV_ + tx];
    __syncthreads();
    _Float16* dst = Vt + ((size_t)bh * DV_ + d0) * S_ + s0;
    #pragma unroll
    for (int i = 0; i < 4; ++i)
        dst[(size_t)(ty + 8 * i) * S_ + tx] = (_Float16)tile[tx][ty + 8 * i];
}

// ======================= main attention kernel =======================
__global__ __launch_bounds__(256)
void fa_fwd_f16wmma_tdm_kernel(const float* __restrict__ Q,
                               const _Float16* __restrict__ Kh,  // [bh][s][dk] f16
                               const _Float16* __restrict__ Vth, // [bh][dv][s] f16
                               float* __restrict__ O) {
    // Double-buffered f16 K / V^T tiles staged by the TDM; per-wave P scratch.
    __shared__ __attribute__((aligned(32))) _Float16 Klds[2][KCH][DK_];  // 16 KB
    __shared__ __attribute__((aligned(32))) _Float16 Vtlds[2][DV_][KCH]; // 16 KB
    __shared__ __attribute__((aligned(16))) float    Plds[NWAVES][16 * KCH]; // 16 KB

    const int tid  = threadIdx.x;
    const int lane = tid & 31;
    const int wid  = tid >> 5;
    const int lh   = lane >> 4;    // half-wave (lanes 16-31)
    const int ln   = lane & 15;

    const int bh = blockIdx.y;             // b*H + h
    const int q0 = blockIdx.x * QTILE;     // block's first Q row
    const int qw = q0 + wid * 16;          // wave's first Q row

    const float*    Qbh = Q   + (size_t)bh * S_ * DK_;
    const _Float16* Kbh = Kh  + (size_t)bh * S_ * DK_;
    const _Float16* Vbh = Vth + (size_t)bh * DV_ * S_;
    float*          Obh = O   + (size_t)bh * S_ * DV_;

    // ---- Q fragments in WMMA A-layout (16-bit A 16x32), 1/sqrt(dk) folded in ----
    // lanes 0-15 row M=lane; contiguous k (this code's order, not ISA table): VGPR v -> K = 2v + 16*lh
    const float qscale = 0.08838834764831845f; // 1/sqrt(128)
    v16h qfrag[4];
    {
        const float* qrow = Qbh + (size_t)(qw + ln) * DK_;
        #pragma unroll
        for (int c = 0; c < 4; ++c) {
            #pragma unroll
            for (int v = 0; v < 8; ++v) {
                int k0 = 32 * c + 2 * v + 16 * lh;
                qfrag[c][2 * v]     = (_Float16)(qrow[k0] * qscale);
                qfrag[c][2 * v + 1] = (_Float16)(qrow[k0 + 1] * qscale);
            }
        }
    }

    // ---- accumulators: O 16x128 as 8 f32 C-tiles, online-softmax row state ----
    v8f o[8] = {};
    float mrow[8], lrow[8];
    #pragma unroll
    for (int r = 0; r < 8; ++r) { mrow[r] = -1e30f; lrow[r] = 0.0f; }

    const int nchunks = (q0 + QTILE) / KCH;   // causal: only chunks touching rows

    // ---- prologue: TDM-stage chunk 0 into buffer 0 (one wave issues the DMA) ----
    if (wid == 0) {
        // K tile: 128 x 32 rows of f16, row stride 128
        tdm_load_2d((unsigned)(size_t)&Klds[0][0][0], Kbh,
                    1u, DK_, S_, DK_, KCH, DK_);
        // V^T tile: 32 cols x 128 rows of f16 out of [DV][S], row stride S
        tdm_load_2d((unsigned)(size_t)&Vtlds[0][0][0], Vbh,
                    1u, S_, DV_, KCH, DV_, S_);
    }

    for (int j = 0; j < nchunks; ++j) {
        const int kv0 = j * KCH;
        const int buf = j & 1;

        // issuing wave drains this chunk's DMAs, then barrier publishes LDS to all
        if (wid == 0) __builtin_amdgcn_s_wait_tensorcnt(0);
        __syncthreads();
        // Everyone has finished reading buffer (j+1)&1 (used in chunk j-1), so it
        // is safe to start the next chunk's DMA now; it overlaps chunk j compute.
        if (wid == 0 && (j + 1) < nchunks) {
            const int nb = (j + 1) & 1;
            tdm_load_2d((unsigned)(size_t)&Klds[nb][0][0],
                        Kbh + (size_t)(kv0 + KCH) * DK_,
                        1u, DK_, S_, DK_, KCH, DK_);
            tdm_load_2d((unsigned)(size_t)&Vtlds[nb][0][0],
                        Vbh + (size_t)(kv0 + KCH),
                        1u, S_, DV_, KCH, DV_, S_);
        }

        // wave-uniform causal skip: chunk useful iff some col <= some row
        if (kv0 <= qw + 15) {
            // ---- S = Q * K^T : 16x32 scores as two 16x16 f32 tiles ----
            v8f sa = {}, sb = {};
            #pragma unroll
            for (int c = 0; c < 4; ++c) {
                // B 32x16 in this code's contiguous k order (as qfrag): lane col n=ln, elem i -> K = 16*lh + i
                v16h kb0 = *(const v16h*)(&Klds[buf][ln     ][32 * c + 16 * lh]);
                v16h kb1 = *(const v16h*)(&Klds[buf][16 + ln][32 * c + 16 * lh]);
                sa = wmma_f16(qfrag[c], kb0, sa);
                sb = wmma_f16(qfrag[c], kb1, sb);
            }

            // ---- causal mask + online softmax (C-layout: row = r+8*lh, col = ln) ----
            const bool needMask = (kv0 + KCH - 1) > qw;  // wave-uniform
            float alpha[8];
            #pragma unroll
            for (int r = 0; r < 8; ++r) {
                int  row = qw + r + 8 * lh;
                float a = sa[r], b = sb[r];
                if (needMask) {
                    a = ((kv0 + ln)      <= row) ? a : -1e30f;
                    b = ((kv0 + 16 + ln) <= row) ? b : -1e30f;
                }
                float mx   = red_max16(fmaxf(a, b));
                float mnew = fmaxf(mrow[r], mx);
                alpha[r]   = __expf(mrow[r] - mnew);
                mrow[r]    = mnew;
                float pa = __expf(a - mnew);
                float pb = __expf(b - mnew);
                lrow[r] = lrow[r] * alpha[r] + red_sum16(pa + pb);
                // stash probabilities row-major for A-layout repack (same-wave only)
                Plds[wid][(r + 8 * lh) * KCH + ln]      = pa;
                Plds[wid][(r + 8 * lh) * KCH + 16 + ln] = pb;
            }

            // rescale running O by alpha (per-row)
            #pragma unroll
            for (int t = 0; t < 8; ++t)
                #pragma unroll
                for (int r = 0; r < 8; ++r)
                    o[t][r] *= alpha[r];

            // ---- repack P (same-wave LDS round trip) into A-layout f16 ----
            v16h pfrag;
            #pragma unroll
            for (int v = 0; v < 8; ++v) {
                int k0 = 2 * v + 16 * lh;
                pfrag[2 * v]     = (_Float16)Plds[wid][ln * KCH + k0];
                pfrag[2 * v + 1] = (_Float16)Plds[wid][ln * KCH + k0 + 1];
            }

            // ---- O += P(16x32) * V(32x128): 8 dv-tiles, k=32 in one WMMA each ----
            #pragma unroll
            for (int t = 0; t < 8; ++t) {
                // B from V^T, same contiguous k as pfrag: lane col n = 16t+ln, elem i -> kv row 16*lh+i
                v16h vb = *(const v16h*)(&Vtlds[buf][16 * t + ln][16 * lh]);
                o[t] = wmma_f16(pfrag, vb, o[t]);
            }
        }
        // next iteration's tensorcnt-wait + barrier provide all needed ordering
    }

    // ---- normalize and write out (f32) ----
    float inv[8];
    #pragma unroll
    for (int r = 0; r < 8; ++r) inv[r] = 1.0f / lrow[r];
    #pragma unroll
    for (int t = 0; t < 8; ++t) {
        #pragma unroll
        for (int r = 0; r < 8; ++r) {
            Obh[(size_t)(qw + r + 8 * lh) * DV_ + 16 * t + ln] = o[t][r] * inv[r];
        }
    }
}

extern "C" void kernel_launch(void* const* d_in, const int* in_sizes, int n_in,
                              void* d_out, int out_size, void* d_ws, size_t ws_size,
                              hipStream_t stream) {
    (void)in_sizes; (void)n_in; (void)ws_size; (void)out_size;
    const float* Q = (const float*)d_in[0];
    const float* K = (const float*)d_in[1];
    const float* V = (const float*)d_in[2];
    // d_in[3] (mask) unused: causality implemented analytically in-kernel
    float* O = (float*)d_out;

    // workspace: f16 K (32 MB) + f16 V-transposed (32 MB)
    const size_t nelem = (size_t)B_ * H_ * S_ * DK_;
    _Float16* Kh  = (_Float16*)d_ws;
    _Float16* Vth = (_Float16*)((char*)d_ws + nelem * sizeof(_Float16));

    // prepass: convert K, convert+transpose V (deterministic, rewritten each call)
    cvt_k_f16_kernel<<<(unsigned)(nelem / 4 / 256), 256, 0, stream>>>(K, Kh);
    transpose_v_f16_kernel<<<dim3(S_ / 32, DV_ / 32, B_ * H_), dim3(32, 8), 0, stream>>>(V, Vth);

    // main flash-attention kernel
    dim3 grid(S_ / QTILE, B_ * H_);
    fa_fwd_f16wmma_tdm_kernel<<<grid, 256, 0, stream>>>(Q, Kh, Vth, O);
}
